// Model_62783831933057
// MI455X (gfx1250) — matrix operand order repaired and checked by reading, not compiled, not run on hardware
//
#include <hip/hip_runtime.h>
#include <cstdint>
#include <cstddef>

// ---------------------------------------------------------------- types ----
typedef __attribute__((ext_vector_type(16))) __bf16 v16bf;
typedef __attribute__((ext_vector_type(8)))  float  v8f;
typedef __attribute__((ext_vector_type(4)))  unsigned int v4u;
typedef __attribute__((ext_vector_type(4)))  int v4i;
typedef unsigned short u16;

union ABu { v4u u[2]; v16bf v; };

#define B_  16
#define S_  1024
#define D_  512
#define H_  1024
#define M_  (B_ * S_)           // 16384 token rows
#define LOG2GAMMA (-0.04580369f) // log2(1 - 2^-5)
#define KSCALE 0.04419417382415922f // 512^-0.5

__device__ __forceinline__ u16 f2bf(float f) {
  unsigned int u = __float_as_uint(f);
  u += 0x7FFFu + ((u >> 16) & 1u);   // round-to-nearest-even
  return (u16)(u >> 16);
}
__device__ __forceinline__ float geluf(float x) { // tanh approximation (jax default)
  float x3 = x * x * x;
  return 0.5f * x * (1.0f + tanhf(0.7978845608028654f * (x + 0.044715f * x3)));
}
__device__ __forceinline__ float siluf(float x) { return x / (1.0f + expf(-x)); }

// ---------------- CDNA5 async global->LDS (ASYNCcnt-tracked) helpers ----
#if defined(__has_builtin) && __has_builtin(__builtin_amdgcn_global_load_async_to_lds_b128)
#define HAVE_ASYNC_LDS 1
#else
#define HAVE_ASYNC_LDS 0
#endif

typedef __attribute__((address_space(1))) v4i* gv4i_p;   // global 16B vector ptr
typedef __attribute__((address_space(3))) v4i* lv4i_p;   // LDS 16B vector ptr

__device__ __forceinline__ void async_load_b128(const u16* gsrc, u16* lds_dst) {
#if HAVE_ASYNC_LDS
  __builtin_amdgcn_global_load_async_to_lds_b128(
      (gv4i_p)(uintptr_t)gsrc, (lv4i_p)(uintptr_t)lds_dst, 0, 0);
#else
  *(v4u*)lds_dst = *(const v4u*)gsrc;   // synchronous fallback
#endif
}

__device__ __forceinline__ void wait_async0() {
#if HAVE_ASYNC_LDS
#if __has_builtin(__builtin_amdgcn_s_wait_asynccnt)
  __builtin_amdgcn_s_wait_asynccnt(0);
#else
  asm volatile("s_wait_asynccnt 0x0" ::: "memory");
#endif
#endif
}

// ---------------------------------------------- weight convert + transpose ----
// W [K,N] fp32  ->  Wt [N,K] bf16   (Wt serves as WMMA B operand: K-contiguous)
__global__ __launch_bounds__(256) void prep_wT_kernel(const float* __restrict__ W,
                                                      u16* __restrict__ Wt,
                                                      int K, int N) {
  int i = blockIdx.x * 256 + threadIdx.x;
  if (i < K * N) {
    int n  = i / K;
    int kk = i - n * K;
    Wt[i] = f2bf(W[(size_t)kk * N + n]);
  }
}

// ------------------------------------------------------------ layernorm ----
// one block per row of 512; fp32 in -> bf16 out
__global__ __launch_bounds__(256) void layernorm_kernel(const float* __restrict__ x,
                                                        const float* __restrict__ g,
                                                        const float* __restrict__ be,
                                                        u16* __restrict__ out) {
  __shared__ float s1[256], s2[256];
  int t = threadIdx.x;
  size_t base = (size_t)blockIdx.x * D_;
  float a = x[base + t], b = x[base + t + 256];
  s1[t] = a + b;
  s2[t] = a * a + b * b;
  __syncthreads();
  for (int st = 128; st > 0; st >>= 1) {
    if (t < st) { s1[t] += s1[t + st]; s2[t] += s2[t + st]; }
    __syncthreads();
  }
  float mean = s1[0] * (1.0f / D_);
  float var  = s2[0] * (1.0f / D_) - mean * mean;
  float rs   = rsqrtf(var + 1e-6f);
  out[base + t]       = f2bf((a - mean) * rs * g[t] + be[t]);
  out[base + t + 256] = f2bf((b - mean) * rs * g[t + 256] + be[t + 256]);
}

// ------------------------------------------- groupnorm + silu gate -> bf16 ----
__global__ __launch_bounds__(256) void groupnorm_gate_kernel(const float* __restrict__ ret,
                                                             const float* __restrict__ gate,
                                                             u16* __restrict__ out) {
  __shared__ float s1[256], s2[256];
  int t = threadIdx.x;
  size_t base = (size_t)blockIdx.x * D_;
  float a = ret[base + t], b = ret[base + t + 256];
  s1[t] = a + b;
  s2[t] = a * a + b * b;
  __syncthreads();
  for (int st = 128; st > 0; st >>= 1) {
    if (t < st) { s1[t] += s1[t + st]; s2[t] += s2[t + st]; }
    __syncthreads();
  }
  float mean = s1[0] * (1.0f / D_);
  float var  = s2[0] * (1.0f / D_) - mean * mean;
  float rs   = rsqrtf(var + 1e-6f);
  float g0 = gate[base + t], g1 = gate[base + t + 256];
  out[base + t]       = f2bf(siluf(g0) * ((a - mean) * rs));
  out[base + t + 256] = f2bf(siluf(g1) * ((b - mean) * rs));
}

// -------------------------------------------------------- WMMA bf16 GEMM ----
// C[M,N] = A[M,K](bf16, row-major) x Bt[N,K](bf16, transposed weight)
// block = 8 waves; wave computes 16(M) x 64(N); block tile 128 x 64.
// B tile (64x32 bf16 = 4KB) is staged global->LDS with async-to-LDS loads,
// double-buffered; A streams straight to VGPRs with prefetch hints.
enum { EPI_BF16 = 0, EPI_BF16_T = 1, EPI_F32 = 2, EPI_RESID = 3,
       EPI_BIAS_GELU_BF16 = 4, EPI_BIAS_RESID = 5 };

template <int EPI>
__global__ __launch_bounds__(256) void gemm_bf16_kernel(
    const u16* __restrict__ A, const u16* __restrict__ Bt,
    void* __restrict__ Cout, const float* __restrict__ bias,
    const float* __restrict__ resid, int N, int K, float scale) {
  __shared__ u16 sB[2][64][32];
  const int tid  = threadIdx.x;
  const int lane = tid & 31;
  const int w    = tid >> 5;
  const int hi   = lane >> 4;       // half-wave select
  const int lo   = lane & 15;
  const int m0   = blockIdx.x * 128 + w * 16;
  const int n0   = blockIdx.y * 64;
  const int kbA  = hi * 8;          // A: lo-half K 0..7 / hi-half K 8..15 (+16 for elems 8..15)

  // async staging role: thread -> one 16B chunk of the 64x32 B tile
  const int bn = tid >> 2;          // local n row 0..63
  const int bs = (tid & 3) * 8;     // element offset in k-slice
  const u16* gB = Bt + (size_t)(n0 + bn) * K + bs;

  async_load_b128(gB, &sB[0][bn][bs]);     // prologue: stage k=0 slice

  v8f acc[4] = {};
  const u16* Arow = A + (size_t)(m0 + lo) * K;
  wait_async0();
  __syncthreads();
  int p = 0;
  for (int k = 0; k < K; k += 32) {
    if (k + 32 < K) {
      async_load_b128(gB + k + 32, &sB[p ^ 1][bn][bs]);   // stream next slice
      __builtin_prefetch(Arow + k + 32, 0, 1);            // global_prefetch next A
    }
    ABu a;
    a.u[0] = *(const v4u*)(Arow + k + kbA);
    a.u[1] = *(const v4u*)(Arow + k + kbA + 16);
#pragma unroll
    for (int t = 0; t < 4; ++t) {
      const u16* lp = &sB[p][t * 16 + lo][hi * 8];
      ABu b;
      b.u[0] = *(const v4u*)(lp);
      b.u[1] = *(const v4u*)(lp + 16);
      acc[t] = __builtin_amdgcn_wmma_f32_16x16x32_bf16(false, a.v, false, b.v,
                                                       (short)0, acc[t], false, false);
    }
    wait_async0();
    __syncthreads();
    p ^= 1;
  }
#pragma unroll
  for (int t = 0; t < 4; ++t) {
    const int n = n0 + t * 16 + lo;
#pragma unroll
    for (int r = 0; r < 8; ++r) {
      const int m = m0 + r + hi * 8;          // C layout: VGPR r -> row r (+8 for hi half)
      float v = acc[t][r] * scale;
      const size_t idx = (size_t)m * N + n;
      if constexpr (EPI == EPI_BF16) {
        ((u16*)Cout)[idx] = f2bf(v);
      } else if constexpr (EPI == EPI_BF16_T) {  // store v^T: [B][D][S]
        const int bb = m >> 10, ss = m & 1023;
        ((u16*)Cout)[(((size_t)bb * D_ + n) << 10) + ss] = f2bf(v);
      } else if constexpr (EPI == EPI_F32) {
        ((float*)Cout)[idx] = v;
      } else if constexpr (EPI == EPI_RESID) {
        ((float*)Cout)[idx] = v + resid[idx];
      } else if constexpr (EPI == EPI_BIAS_GELU_BF16) {
        ((u16*)Cout)[idx] = f2bf(geluf(v + bias[n]));
      } else { // EPI_BIAS_RESID
        ((float*)Cout)[idx] = v + bias[n] + resid[idx];
      }
    }
  }
}

// ------------------------------------------------------------ retention ----
// block: 32 query rows (2 sub-tiles of 16), 8 waves.
// wave w: qi = w>>2 selects query sub-tile, dj = w&3 owns 128 output d-cols.
// per 32-key chunk: waves 0..3 compute the two 16x32 score tiles via WMMA over
// d, apply causal decay gamma^(n-m) in fp32, round to bf16 in LDS; then all 8
// waves consume scores (A-layout from LDS) against v^T slabs via WMMA.
// NOTE: score-phase guard uses readfirstlane so it compiles to a *scalar*
// branch (EXEC stays all-ones around every WMMA, per ISA 7.12 restrictions).
__global__ __launch_bounds__(256) void retention_kernel(
    const u16* __restrict__ q, const u16* __restrict__ k,
    const u16* __restrict__ vT, float* __restrict__ ret) {
  __shared__ u16 s_sc[2][16][32];
  const int lane = threadIdx.x & 31;
  const int w    = threadIdx.x >> 5;
  const int wu   = __builtin_amdgcn_readfirstlane(w);   // wave-uniform scalar copy
  const int hi   = lane >> 4;
  const int lo   = lane & 15;
  const int b    = blockIdx.y;
  const int q0   = blockIdx.x * 32;
  const int qi   = w >> 2;
  const int dj   = w & 3;
  const u16* qb = q  + (size_t)b * S_ * D_;
  const u16* kb = k  + (size_t)b * S_ * D_;
  const u16* vb = vT + (size_t)b * D_ * S_;

  // score-wave role (only meaningful for wu < 4)
  const int qis  = w >> 1;
  const int half = w & 1;
  const int qr0s = q0 + qis * 16;

  v8f acc[8] = {};
  const int nchunk = blockIdx.x + 1;   // key chunks 0..q-block (causal)
  for (int c = 0; c < nchunk; ++c) {
    const int m0 = c * 32;
    if (wu < 4) {                      // scalar branch: EXEC untouched
      v8f sc = {};
      const u16* Ap = qb + (size_t)(qr0s + lo) * D_;
      const u16* Bp = kb + (size_t)(m0 + half * 16 + lo) * D_ + hi * 8;
      const int kbA = hi * 8;
      for (int kd = 0; kd < D_; kd += 32) {
        ABu a, bb;
        a.u[0]  = *(const v4u*)(Ap + kd + kbA);
        a.u[1]  = *(const v4u*)(Ap + kd + kbA + 16);
        bb.u[0] = *(const v4u*)(Bp + kd);
        bb.u[1] = *(const v4u*)(Bp + kd + 16);
        sc = __builtin_amdgcn_wmma_f32_16x16x32_bf16(false, a.v, false, bb.v,
                                                     (short)0, sc, false, false);
      }
      const int kcol = m0 + half * 16 + lo;
#pragma unroll
      for (int r = 0; r < 8; ++r) {
        const int qrow = qr0s + r + hi * 8;
        const int dt = qrow - kcol;
        float s = (dt >= 0) ? sc[r] * exp2f((float)dt * LOG2GAMMA) : 0.0f;
        s_sc[qis][r + hi * 8][half * 16 + lo] = f2bf(s);
      }
    }
    __syncthreads();
    ABu a;                              // scores in A-layout from LDS
    a.u[0] = *(const v4u*)&s_sc[qi][lo][hi * 8];
    a.u[1] = *(const v4u*)&s_sc[qi][lo][16 + hi * 8];
#pragma unroll
    for (int t = 0; t < 8; ++t) {
      const u16* Bp = vb + (size_t)(dj * 128 + t * 16 + lo) * S_ + m0 + hi * 8;
      ABu bb;
      bb.u[0] = *(const v4u*)(Bp);
      bb.u[1] = *(const v4u*)(Bp + 16);
      acc[t] = __builtin_amdgcn_wmma_f32_16x16x32_bf16(false, a.v, false, bb.v,
                                                       (short)0, acc[t], false, false);
    }
    __syncthreads();
  }
  float* rout = ret + (size_t)b * S_ * D_;
#pragma unroll
  for (int t = 0; t < 8; ++t) {
    const int n = dj * 128 + t * 16 + lo;
#pragma unroll
    for (int r = 0; r < 8; ++r) {
      const int m = q0 + qi * 16 + r + hi * 8;
      rout[(size_t)m * D_ + n] = acc[t][r];
    }
  }
}

// ------------------------------------------------------- pool + readout ----
__global__ __launch_bounds__(256) void pool_kernel(const float* __restrict__ h,
                                                   float* __restrict__ pooled) {
  int b = blockIdx.x, t = threadIdx.x;
  const float* hb = h + (size_t)b * S_ * D_;
  float s0 = 0.f, s1 = 0.f;
  for (int s = 0; s < S_; ++s) {
    s0 += hb[(size_t)s * D_ + t];
    s1 += hb[(size_t)s * D_ + t + 256];
  }
  pooled[b * D_ + t] = s0;
  pooled[b * D_ + t + 256] = s1;
}

__global__ void readout_kernel(const float* __restrict__ pooled,
                               const float* __restrict__ Wro,
                               const float* __restrict__ bro,
                               float* __restrict__ out) {
  int t = threadIdx.x;
  if (t >= 32) return;
  int b = t >> 1, o = t & 1;
  float s = bro[o];
  for (int j = 0; j < D_; ++j) s += pooled[b * D_ + j] * Wro[j * 2 + o];
  out[t] = s;   // t == b*2 + o
}

// -------------------------------------------------------------- launch ----
extern "C" void kernel_launch(void* const* d_in, const int* in_sizes, int n_in,
                              void* d_out, int out_size, void* d_ws, size_t ws_size,
                              hipStream_t stream) {
  (void)in_sizes; (void)n_in; (void)out_size; (void)ws_size;
  const float* x    = (const float*)d_in[0];
  const float* Wq   = (const float*)d_in[1];
  const float* Wk   = (const float*)d_in[2];
  const float* Wv   = (const float*)d_in[3];
  const float* Wg   = (const float*)d_in[4];
  const float* Wo   = (const float*)d_in[5];
  const float* ln1g = (const float*)d_in[6];
  const float* ln1b = (const float*)d_in[7];
  const float* ln2g = (const float*)d_in[8];
  const float* ln2b = (const float*)d_in[9];
  const float* W1   = (const float*)d_in[10];
  const float* b1   = (const float*)d_in[11];
  const float* W2   = (const float*)d_in[12];
  const float* b2   = (const float*)d_in[13];
  const float* Wro  = (const float*)d_in[14];
  const float* bro  = (const float*)d_in[15];

  char* ws = (char*)d_ws;
  size_t off = 0;
  auto take = [&](size_t bytes) {
    char* p = ws + off;
    off += (bytes + 255) & ~(size_t)255;
    return p;
  };
  float* h     = (float*)take((size_t)M_ * D_ * 4);   // hidden state (fp32 spine)
  u16*   xn    = (u16*)  take((size_t)M_ * D_ * 2);   // layernorm output (bf16)
  u16*   qbuf  = (u16*)  take((size_t)M_ * D_ * 2);   // q (later reused as gated)
  u16*   kbuf  = (u16*)  take((size_t)M_ * D_ * 2);   // k (pre-scaled)
  u16*   vTbuf = (u16*)  take((size_t)M_ * D_ * 2);   // v^T [B,D,S]
  char*  gscr  =         take((size_t)M_ * D_ * 4);   // gate fp32 / ffn bf16 (reused)
  float* gate  = (float*)gscr;
  u16*   ffn   = (u16*)gscr;
  float* retb  = (float*)take((size_t)M_ * D_ * 4);
  float* pooled= (float*)take((size_t)B_ * D_ * 4);
  const size_t LSTR = 5 * (size_t)D_ * D_ + 2 * (size_t)D_ * H_; // bf16 elems/layer
  u16*   wT    = (u16*)  take(2 * LSTR * 2);

  // --- weight prep: bf16 + transpose ---
  for (int l = 0; l < 2; ++l) {
    u16* base = wT + l * LSTR;
    prep_wT_kernel<<<1024, 256, 0, stream>>>(Wq + (size_t)l * D_ * D_, base,                 D_, D_);
    prep_wT_kernel<<<1024, 256, 0, stream>>>(Wk + (size_t)l * D_ * D_, base + 1 * D_ * D_,   D_, D_);
    prep_wT_kernel<<<1024, 256, 0, stream>>>(Wv + (size_t)l * D_ * D_, base + 2 * D_ * D_,   D_, D_);
    prep_wT_kernel<<<1024, 256, 0, stream>>>(Wg + (size_t)l * D_ * D_, base + 3 * D_ * D_,   D_, D_);
    prep_wT_kernel<<<1024, 256, 0, stream>>>(Wo + (size_t)l * D_ * D_, base + 4 * D_ * D_,   D_, D_);
    prep_wT_kernel<<<2048, 256, 0, stream>>>(W1 + (size_t)l * D_ * H_, base + 5 * D_ * D_,   D_, H_);
    prep_wT_kernel<<<2048, 256, 0, stream>>>(W2 + (size_t)l * H_ * D_, base + 5 * D_ * D_ + (size_t)D_ * H_, H_, D_);
  }

  (void)hipMemcpyAsync(h, x, (size_t)M_ * D_ * 4, hipMemcpyDeviceToDevice, stream);

  dim3 g512(M_ / 128, D_ / 64);   // N = 512
  dim3 g1024(M_ / 128, H_ / 64);  // N = 1024
  for (int l = 0; l < 2; ++l) {
    u16* base = wT + l * LSTR;
    u16 *wqT = base, *wkT = base + D_ * D_, *wvT = base + 2 * D_ * D_;
    u16 *wgT = base + 3 * D_ * D_, *woT = base + 4 * D_ * D_;
    u16 *w1T = base + 5 * D_ * D_, *w2T = w1T + (size_t)D_ * H_;

    layernorm_kernel<<<M_, 256, 0, stream>>>(h, ln1g + l * D_, ln1b + l * D_, xn);
    gemm_bf16_kernel<EPI_BF16  ><<<g512, 256, 0, stream>>>(xn, wqT, qbuf,  nullptr, nullptr, D_, D_, 1.0f);
    gemm_bf16_kernel<EPI_BF16  ><<<g512, 256, 0, stream>>>(xn, wkT, kbuf,  nullptr, nullptr, D_, D_, KSCALE);
    gemm_bf16_kernel<EPI_BF16_T><<<g512, 256, 0, stream>>>(xn, wvT, vTbuf, nullptr, nullptr, D_, D_, 1.0f);
    gemm_bf16_kernel<EPI_F32   ><<<g512, 256, 0, stream>>>(xn, wgT, gate,  nullptr, nullptr, D_, D_, 1.0f);
    retention_kernel<<<dim3(S_ / 32, B_), 256, 0, stream>>>(qbuf, kbuf, vTbuf, retb);
    groupnorm_gate_kernel<<<M_, 256, 0, stream>>>(retb, gate, qbuf /* reused as gated */);
    gemm_bf16_kernel<EPI_RESID ><<<g512, 256, 0, stream>>>(qbuf, woT, h, nullptr, h, D_, D_, 1.0f);
    layernorm_kernel<<<M_, 256, 0, stream>>>(h, ln2g + l * D_, ln2b + l * D_, xn);
    gemm_bf16_kernel<EPI_BIAS_GELU_BF16><<<g1024, 256, 0, stream>>>(xn, w1T, ffn, b1 + l * H_, nullptr, H_, D_, 1.0f);
    gemm_bf16_kernel<EPI_BIAS_RESID    ><<<g512, 256, 0, stream>>>(ffn, w2T, h, b2 + l * D_, h, D_, H_, 1.0f);
  }

  pool_kernel<<<B_, 256, 0, stream>>>(h, pooled);
  readout_kernel<<<1, 32, 0, stream>>>(pooled, Wro, bro, (float*)d_out);
}
